// StaticEncoder_22832046145627
// MI455X (gfx1250) — matrix operand order repaired and checked by reading, not compiled, not run on hardware
//
#include <hip/hip_runtime.h>
#include <hip/hip_bf16.h>
#include <math.h>

#define N_NODES   20000
#define N_EDGES   320000
#define N_GRAPHS  8
#define NFD       128
#define HID       64
#define OUT_DIM   768
#define HEADS     4

typedef __attribute__((ext_vector_type(16))) __bf16 v16bf;
typedef __attribute__((ext_vector_type(8)))  float  v8f;

__device__ __forceinline__ unsigned short bfbits(float f) {
    union { float f; unsigned u; } a; a.f = f;
    return (unsigned short)((a.u + 0x7FFFu + ((a.u >> 16) & 1u)) >> 16);  // RNE
}

// ordered-float <-> uint key for atomicMax-based segment max
__device__ __forceinline__ unsigned fkey(float f) {
    union { float f; unsigned u; } a; a.f = f;
    return (a.u & 0x80000000u) ? ~a.u : (a.u | 0x80000000u);
}
__device__ __forceinline__ float unfkey(unsigned k) {
    union { unsigned u; float f; } a;
    a.u = (k & 0x80000000u) ? (k ^ 0x80000000u) : ~k;
    return a.f;
}

// ---------------------------------------------------------------------------
// f32 -> bf16 elementwise convert (paired stores), n must be even
// ---------------------------------------------------------------------------
__global__ void cvt_bf16_kernel(const float* __restrict__ in, unsigned* __restrict__ out, size_t n2)
{
    size_t i = (size_t)blockIdx.x * blockDim.x + threadIdx.x;
    size_t st = (size_t)gridDim.x * blockDim.x;
    for (; i < n2; i += st) {
        unsigned lo = bfbits(in[2 * i]);
        unsigned hi = bfbits(in[2 * i + 1]);
        out[i] = lo | (hi << 16);
    }
}

// ---------------------------------------------------------------------------
// Pack weight matrix B[K,N] (f32 row-major) into bf16 fragment-native layout:
// for each (tile16 = n/16, kstep = k/32): 512 bf16, lane-major:
//   elem i of lane L = B[kstep*32 + (L>=16?16:0) + i][tile16*16 + (L&15)]
// one thread per (tile16, kstep, lane) writes 16 contiguous bf16 (32B)
// ---------------------------------------------------------------------------
__global__ void pack_b_kernel(const float* __restrict__ B, unsigned short* __restrict__ bp,
                              int K, int N)
{
    int total = (N >> 4) * (K >> 5) * 32;
    int t = blockIdx.x * blockDim.x + threadIdx.x;
    if (t >= total) return;
    int lane  = t & 31;
    int rest  = t >> 5;
    int ksteps = K >> 5;
    int kstep = rest % ksteps;
    int tile  = rest / ksteps;
    int col = (tile << 4) + (lane & 15);
    int kb  = (kstep << 5) + ((lane >> 4) << 4);
    unsigned short* o = bp + (size_t)t * 16;
#pragma unroll
    for (int i = 0; i < 16; ++i)
        o[i] = bfbits(B[(size_t)(kb + i) * N + col]);
}

// ---------------------------------------------------------------------------
// WMMA bf16 GEMM: C[M,N] = A[M,K] @ B[K,N]; A pre-converted bf16 row-major,
// B pre-packed (above). One 16x32 output per wave (two accumulators sharing
// the A fragment); 8 waves/block. M%16==0, K%32==0, N%32==0.
// ---------------------------------------------------------------------------
__global__ __launch_bounds__(256) void gemm_bf16_wmma(
    const unsigned short* __restrict__ Abf, const unsigned short* __restrict__ Bp,
    float* __restrict__ C, int M, int K, int N)
{
    int wave = blockIdx.x * 8 + (threadIdx.x >> 5);
    int lane = threadIdx.x & 31;
    int tilesN2 = N >> 5;
    int numW = (M >> 4) * tilesN2;
    if (wave >= numW) return;            // whole-wave uniform: EXEC stays all-ones

    int tm   = (wave / tilesN2) << 4;
    int tn2  = (wave % tilesN2);         // covers 16-col tiles 2*tn2, 2*tn2+1
    int half = lane >> 4;
    int l16  = lane & 15;
    int ksteps = K >> 5;

    const uint4* ap  = (const uint4*)(Abf + (size_t)(tm + l16) * K + (half << 4));
    const uint4* bp0 = (const uint4*)(Bp + (size_t)(2 * tn2)     * ksteps * 512) + lane * 2;
    const uint4* bp1 = (const uint4*)(Bp + (size_t)(2 * tn2 + 1) * ksteps * 512) + lane * 2;

    v8f acc0 = {0.f, 0.f, 0.f, 0.f, 0.f, 0.f, 0.f, 0.f};
    v8f acc1 = {0.f, 0.f, 0.f, 0.f, 0.f, 0.f, 0.f, 0.f};

#pragma unroll 2
    for (int ks = 0; ks < ksteps; ++ks) {
        __builtin_prefetch(ap + 8, 0, 3);          // next-next A fragment
        union { uint4 q[2]; v16bf v; } a, b0, b1;
        a.q[0]  = ap[0];  a.q[1]  = ap[1];          // elems 0..7, 8..15 (K +8)
        b0.q[0] = bp0[0]; b0.q[1] = bp0[1];
        b1.q[0] = bp1[0]; b1.q[1] = bp1[1];
        acc0 = __builtin_amdgcn_wmma_f32_16x16x32_bf16(false, a.v, false, b0.v,
                                                       (short)0, acc0, false, false);
        acc1 = __builtin_amdgcn_wmma_f32_16x16x32_bf16(false, a.v, false, b1.v,
                                                       (short)0, acc1, false, false);
        ap  += 4;      // 32 bf16 = 4 x uint4
        bp0 += 64;     // 512 bf16 = 64 x uint4
        bp1 += 64;
    }

    int r0 = tm + (half << 3);
    float* c0 = C + (size_t)r0 * N + ((tn2 << 5) + l16);
    float* c1 = c0 + 16;
#pragma unroll
    for (int i = 0; i < 8; ++i) {
        c0[(size_t)i * N] = acc0[i];
        c1[(size_t)i * N] = acc1[i];
    }
}

// ---------------------------------------------------------------------------
// Node encoder
// ---------------------------------------------------------------------------
__global__ void enc1_kernel(const float* __restrict__ x, const float* __restrict__ w,
                            const float* __restrict__ b, float* __restrict__ o)
{
    int t = blockIdx.x * blockDim.x + threadIdx.x;
    if (t >= N_NODES * 64) return;
    int n = t >> 6, j = t & 63;
    const float* xr = x + (size_t)n * 5;
    float s = b[j];
#pragma unroll
    for (int k = 0; k < 5; ++k) s += xr[k] * w[k * 64 + j];
    o[t] = s > 0.f ? s : 0.f;
}

__global__ void enc2_kernel(const float* __restrict__ h, const float* __restrict__ w,
                            const float* __restrict__ b, float* __restrict__ o)
{
    int t = blockIdx.x * blockDim.x + threadIdx.x;
    if (t >= N_NODES * NFD) return;
    int n = t >> 7, j = t & 127;
    const float* hr = h + (size_t)n * 64;
    float s = b[j];
    for (int k = 0; k < 64; ++k) s += hr[k] * w[k * NFD + j];
    o[t] = s > 0.f ? s : 0.f;
}

// ---------------------------------------------------------------------------
// GAT helper kernels
// ---------------------------------------------------------------------------
__global__ void attn_coef_kernel(const float* __restrict__ hf,
                                 const float* __restrict__ asrc, const float* __restrict__ adst,
                                 float* __restrict__ a_s, float* __restrict__ a_d, int H, int C)
{
    int t = blockIdx.x * blockDim.x + threadIdx.x;
    if (t >= N_NODES * H) return;
    int n = t / H, h = t % H;
    const float* row = hf + (size_t)n * H * C + (size_t)h * C;
    float s1 = 0.f, s2 = 0.f;
    for (int c = 0; c < C; ++c) {
        float v = row[c];
        s1 += v * asrc[h * C + c];
        s2 += v * adst[h * C + c];
    }
    a_s[t] = s1; a_d[t] = s2;
}

__global__ void easum_kernel(const float* __restrict__ ea, float* __restrict__ sums)
{
    __shared__ float lds[256];
    int tid = threadIdx.x;
    float local = 0.f;
    for (size_t i = (size_t)blockIdx.x * 256 + tid; i < (size_t)N_EDGES * 4;
         i += (size_t)gridDim.x * 256)
        local += ea[i];                       // i & 3 == tid & 3 (strides multiple of 4)
    lds[tid] = local; __syncthreads();
    for (int s = 128; s >= 4; s >>= 1) {
        if (tid < s) lds[tid] += lds[tid + s];
        __syncthreads();
    }
    if (tid < 4) atomicAdd(&sums[tid], lds[tid]);
}

// w_eff[k,h] = sum_c We[k, h*C+c] * att_edge[h,c];  a_e_loop[h] = mean(ea) . w_eff[:,h]
__global__ void weff_kernel(const float* __restrict__ We, const float* __restrict__ aedge,
                            const float* __restrict__ easum,
                            float* __restrict__ weff, float* __restrict__ aeloop, int H, int C)
{
    __shared__ float wl[4 * HEADS];
    int t = threadIdx.x;
    if (t < 4 * H) {
        int k = t / H, h = t % H;
        float s = 0.f;
        for (int c = 0; c < C; ++c) s += We[(size_t)k * H * C + h * C + c] * aedge[h * C + c];
        wl[t] = s; weff[t] = s;
    }
    __syncthreads();
    if (t < H) {
        float s = 0.f;
        const float inv = 1.0f / (float)N_EDGES;
#pragma unroll
        for (int k = 0; k < 4; ++k) s += (easum[k] * inv) * wl[k * H + t];
        aeloop[t] = s;
    }
}

// self-loop logit; init segment-max keys with it (self loop always present); zero z
__global__ void node_init_kernel(const float* __restrict__ a_s, const float* __restrict__ a_d,
                                 const float* __restrict__ aeloop,
                                 float* __restrict__ llog, unsigned* __restrict__ mkey,
                                 float* __restrict__ z, int H)
{
    int t = blockIdx.x * blockDim.x + threadIdx.x;
    if (t >= N_NODES * H) return;
    int h = t % H;
    float l = a_s[t] + a_d[t] + aeloop[h];
    l = l > 0.f ? l : 0.2f * l;
    llog[t] = l;
    mkey[t] = fkey(l);
    z[t] = 0.f;
}

__global__ void edge_max_kernel(const int* __restrict__ src, const int* __restrict__ dst,
                                const float* __restrict__ ea,
                                const float* __restrict__ a_s, const float* __restrict__ a_d,
                                const float* __restrict__ weff,
                                float* __restrict__ elog, unsigned* __restrict__ mkey, int H)
{
    int t = blockIdx.x * blockDim.x + threadIdx.x;
    if (t >= N_EDGES * H) return;
    int e = t / H, h = t % H;
    int s = src[e], d = dst[e];
    float aev = 0.f;
#pragma unroll
    for (int k = 0; k < 4; ++k) aev += ea[(size_t)e * 4 + k] * weff[k * H + h];
    float l = a_s[s * H + h] + a_d[d * H + h] + aev;
    l = l > 0.f ? l : 0.2f * l;
    elog[t] = l;
    atomicMax(&mkey[d * H + h], fkey(l));
}

__global__ void edge_exp_kernel(const int* __restrict__ dst, const unsigned* __restrict__ mkey,
                                float* __restrict__ ew, float* __restrict__ z, int H)
{
    int t = blockIdx.x * blockDim.x + threadIdx.x;
    if (t >= N_EDGES * H) return;
    int e = t / H, h = t % H;
    int d = dst[e];
    float m = unfkey(mkey[d * H + h]);
    float w = expf(ew[t] - m);       // ew currently holds the logit
    ew[t] = w;
    atomicAdd(&z[d * H + h], w);
}

__global__ void node_exp_kernel(const unsigned* __restrict__ mkey,
                                float* __restrict__ llog, float* __restrict__ z, int H)
{
    int t = blockIdx.x * blockDim.x + threadIdx.x;
    if (t >= N_NODES * H) return;
    float w = expf(llog[t] - unfkey(mkey[t]));
    llog[t] = w;                     // now holds self-loop weight
    atomicAdd(&z[t], w);
}

__global__ void zero_kernel(float* __restrict__ p, size_t n)
{
    size_t i = (size_t)blockIdx.x * blockDim.x + threadIdx.x;
    size_t st = (size_t)gridDim.x * blockDim.x;
    for (; i < n; i += st) p[i] = 0.f;
}

// one wave per edge, lanes across channels (coalesced gather + atomic scatter)
__global__ __launch_bounds__(256) void edge_agg_kernel(
    const int* __restrict__ src, const int* __restrict__ dst,
    const float* __restrict__ ew, const float* __restrict__ z,
    const float* __restrict__ hf, float* __restrict__ acc, int H, int C)
{
    int e = blockIdx.x * 8 + (threadIdx.x >> 5);
    if (e >= N_EDGES) return;
    int lane = threadIdx.x & 31;
    int s = src[e], d = dst[e];
    int HC = H * C;
    const float* hs = hf + (size_t)s * HC;
    float* od = acc + (size_t)d * HC;
    for (int c = lane; c < HC; c += 32) {
        int h = c / C;
        float alpha = ew[(size_t)e * H + h] / (z[(size_t)d * H + h] + 1e-16f);
        atomicAdd(&od[c], alpha * hs[c]);
    }
}

__global__ void finalize_kernel(const float* __restrict__ acc, const float* __restrict__ hf,
                                const float* __restrict__ wloop, const float* __restrict__ z,
                                const float* __restrict__ bias, float* __restrict__ out,
                                int H, int C, int do_elu)
{
    size_t t = (size_t)blockIdx.x * blockDim.x + threadIdx.x;
    int HC = H * C;
    if (t >= (size_t)N_NODES * HC) return;
    size_t n = t / HC;
    int c = (int)(t % HC);
    int h = c / C;
    float alpha = wloop[n * H + h] / (z[n * H + h] + 1e-16f);
    float v = acc[t] + alpha * hf[t] + bias[c];
    if (do_elu) v = v > 0.f ? v : expm1f(v);
    out[t] = v;
}

// ---------------------------------------------------------------------------
// Pool + projection head
// ---------------------------------------------------------------------------
__global__ void pool_sum_kernel(const float* __restrict__ h, const int* __restrict__ batch,
                                float* __restrict__ gsum)
{
    int t = blockIdx.x * blockDim.x + threadIdx.x;
    if (t >= N_NODES * HID) return;
    int n = t >> 6, c = t & 63;
    atomicAdd(&gsum[batch[n] * HID + c], h[t]);
}
__global__ void pool_cnt_kernel(const int* __restrict__ batch, float* __restrict__ cnt)
{
    int n = blockIdx.x * blockDim.x + threadIdx.x;
    if (n >= N_NODES) return;
    atomicAdd(&cnt[batch[n]], 1.0f);
}
__global__ void pool_div_kernel(const float* __restrict__ gsum, const float* __restrict__ cnt,
                                float* __restrict__ g)
{
    int t = blockIdx.x * blockDim.x + threadIdx.x;
    if (t >= N_GRAPHS * HID) return;
    g[t] = gsum[t] / fmaxf(cnt[t >> 6], 1.0f);
}

__global__ void proj1_kernel(const float* __restrict__ g, const float* __restrict__ w,
                             const float* __restrict__ b, float* __restrict__ p1)
{
    int t = blockIdx.x * blockDim.x + threadIdx.x;
    if (t >= N_GRAPHS * 2 * HID) return;
    int bb = t >> 7, j = t & 127;
    float s = b[j];
    for (int k = 0; k < HID; ++k) s += g[bb * HID + k] * w[k * 2 * HID + j];
    p1[t] = s > 0.f ? s : 0.f;
}
__global__ void proj2_kernel(const float* __restrict__ p1, const float* __restrict__ w,
                             const float* __restrict__ b, float* __restrict__ statf)
{
    int t = blockIdx.x * blockDim.x + threadIdx.x;
    if (t >= N_GRAPHS * OUT_DIM) return;
    int bb = t / OUT_DIM, j = t % OUT_DIM;
    float s = b[j];
    for (int k = 0; k < 2 * HID; ++k) s += p1[bb * 2 * HID + k] * w[k * OUT_DIM + j];
    statf[t] = s;
}
__global__ void vuln_kernel(const float* __restrict__ statf, const float* __restrict__ w,
                            const float* __restrict__ b, float* __restrict__ sc)
{
    int t = blockIdx.x * blockDim.x + threadIdx.x;
    if (t >= N_GRAPHS * 5) return;
    int bb = t / 5, j = t % 5;
    float s = b[j];
    for (int k = 0; k < OUT_DIM; ++k) s += statf[bb * OUT_DIM + k] * w[k * 5 + j];
    sc[t] = 1.f / (1.f + expf(-s));
}

// ---------------------------------------------------------------------------
// Host side
// ---------------------------------------------------------------------------
static void run_gat_layer(const float* in, int Kin, int H, int C,
                          const float* Wg, const float* asrc, const float* adst,
                          const float* We, const float* aedge, const float* bias, int do_elu,
                          const int* src, const int* dst, const float* ea,
                          unsigned short* a_bf, unsigned short* b_bf,
                          float* hf, float* out, float* a_s, float* a_d,
                          unsigned* mkey, float* zz, float* llog, float* ebuf,
                          const float* easum, float* weff, float* aeloop, hipStream_t stream)
{
    int HC = H * C;
    // pre-convert A, pre-pack B, then WMMA GEMM
    size_t n2 = (size_t)N_NODES * Kin / 2;
    cvt_bf16_kernel<<<2048, 256, 0, stream>>>(in, (unsigned*)a_bf, n2);
    int packThreads = (HC >> 4) * (Kin >> 5) * 32;
    pack_b_kernel<<<(packThreads + 255) / 256, 256, 0, stream>>>(Wg, b_bf, Kin, HC);
    int numW = (N_NODES / 16) * (HC / 32);
    gemm_bf16_wmma<<<(numW + 7) / 8, 256, 0, stream>>>(a_bf, b_bf, hf, N_NODES, Kin, HC);

    attn_coef_kernel<<<(N_NODES * H + 255) / 256, 256, 0, stream>>>(hf, asrc, adst, a_s, a_d, H, C);
    weff_kernel<<<1, 32, 0, stream>>>(We, aedge, easum, weff, aeloop, H, C);
    node_init_kernel<<<(N_NODES * H + 255) / 256, 256, 0, stream>>>(a_s, a_d, aeloop, llog, mkey, zz, H);
    edge_max_kernel<<<(N_EDGES * H + 255) / 256, 256, 0, stream>>>(src, dst, ea, a_s, a_d, weff, ebuf, mkey, H);
    edge_exp_kernel<<<(N_EDGES * H + 255) / 256, 256, 0, stream>>>(dst, mkey, ebuf, zz, H);
    node_exp_kernel<<<(N_NODES * H + 255) / 256, 256, 0, stream>>>(mkey, llog, zz, H);
    zero_kernel<<<2048, 256, 0, stream>>>(out, (size_t)N_NODES * HC);
    edge_agg_kernel<<<(N_EDGES + 7) / 8, 256, 0, stream>>>(src, dst, ebuf, zz, hf, out, H, C);
    finalize_kernel<<<((size_t)N_NODES * HC + 255) / 256, 256, 0, stream>>>(out, hf, llog, zz, bias, out, H, C, do_elu);
}

extern "C" void kernel_launch(void* const* d_in, const int* in_sizes, int n_in,
                              void* d_out, int out_size, void* d_ws, size_t ws_size,
                              hipStream_t stream)
{
    (void)in_sizes; (void)n_in; (void)out_size; (void)ws_size;
    const float* x       = (const float*)d_in[0];
    const int*   ei      = (const int*)d_in[1];
    const float* ea      = (const float*)d_in[2];
    const int*   batch   = (const int*)d_in[3];
    const float* enc_w1  = (const float*)d_in[4];
    const float* enc_b1  = (const float*)d_in[5];
    const float* enc_w2  = (const float*)d_in[6];
    const float* enc_b2  = (const float*)d_in[7];
    const float* g1_w    = (const float*)d_in[8];
    const float* g1_asrc = (const float*)d_in[9];
    const float* g1_adst = (const float*)d_in[10];
    const float* g1_we   = (const float*)d_in[11];
    const float* g1_aed  = (const float*)d_in[12];
    const float* g1_b    = (const float*)d_in[13];
    const float* g2_w    = (const float*)d_in[14];
    const float* g2_asrc = (const float*)d_in[15];
    const float* g2_adst = (const float*)d_in[16];
    const float* g2_we   = (const float*)d_in[17];
    const float* g2_aed  = (const float*)d_in[18];
    const float* g2_b    = (const float*)d_in[19];
    const float* g3_w    = (const float*)d_in[20];
    const float* g3_asrc = (const float*)d_in[21];
    const float* g3_adst = (const float*)d_in[22];
    const float* g3_we   = (const float*)d_in[23];
    const float* g3_aed  = (const float*)d_in[24];
    const float* g3_b    = (const float*)d_in[25];
    const float* proj_w1 = (const float*)d_in[26];
    const float* proj_b1 = (const float*)d_in[27];
    const float* proj_w2 = (const float*)d_in[28];
    const float* proj_b2 = (const float*)d_in[29];
    const float* vuln_w  = (const float*)d_in[30];
    const float* vuln_b  = (const float*)d_in[31];

    const int* src = ei;
    const int* dst = ei + N_EDGES;

    float* statf  = (float*)d_out;                 // [8,768]
    float* scores = (float*)d_out + N_GRAPHS * OUT_DIM;

    // workspace carve-up (all chunk sizes multiples of 4 floats -> 16B aligned)
    float* W = (float*)d_ws;
    size_t off = 0;
    auto alloc = [&](size_t n) { float* p = W + off; off += n; return p; };
    float* ench  = alloc((size_t)N_NODES * 64);            // encoder hidden
    float* h0    = alloc((size_t)N_NODES * NFD);           // encoder out / layer1 in
    float* bufA  = alloc((size_t)N_NODES * HEADS * HID);   // layer1 out, layer3 out
    float* bufB  = alloc((size_t)N_NODES * HEADS * HID);   // layer2 out
    float* hf    = alloc((size_t)N_NODES * HEADS * HID);   // per-layer h@W
    float* abff  = alloc((size_t)N_NODES * HEADS * HID / 2); // bf16 A (ushort), 16B aligned
    float* bbff  = alloc((size_t)(HEADS * HID) * (HEADS * HID) / 2); // bf16 packed B
    float* a_s   = alloc((size_t)N_NODES * HEADS);
    float* a_d   = alloc((size_t)N_NODES * HEADS);
    float* mkeyf = alloc((size_t)N_NODES * HEADS);
    float* zz    = alloc((size_t)N_NODES * HEADS);
    float* llog  = alloc((size_t)N_NODES * HEADS);
    float* ebuf  = alloc((size_t)N_EDGES * HEADS);         // edge logit -> edge weight
    float* easum = alloc(4);
    float* weff  = alloc(4 * HEADS);
    float* aeloop= alloc(HEADS);
    float* gsum  = alloc(N_GRAPHS * HID);
    float* cnt   = alloc(N_GRAPHS);
    float* g     = alloc(N_GRAPHS * HID);
    float* p1    = alloc(N_GRAPHS * 2 * HID);
    unsigned* mkey = (unsigned*)mkeyf;
    unsigned short* a_bf = (unsigned short*)abff;
    unsigned short* b_bf = (unsigned short*)bbff;

    // edge-attr mean (shared by all layers)
    zero_kernel<<<1, 256, 0, stream>>>(easum, 4);
    easum_kernel<<<256, 256, 0, stream>>>(ea, easum);

    // node encoder
    enc1_kernel<<<(N_NODES * 64 + 255) / 256, 256, 0, stream>>>(x, enc_w1, enc_b1, ench);
    enc2_kernel<<<(N_NODES * NFD + 255) / 256, 256, 0, stream>>>(ench, enc_w2, enc_b2, h0);

    // three GAT layers
    run_gat_layer(h0, NFD, HEADS, HID, g1_w, g1_asrc, g1_adst, g1_we, g1_aed, g1_b, 1,
                  src, dst, ea, a_bf, b_bf, hf, bufA, a_s, a_d, mkey, zz, llog, ebuf,
                  easum, weff, aeloop, stream);
    run_gat_layer(bufA, HEADS * HID, HEADS, HID, g2_w, g2_asrc, g2_adst, g2_we, g2_aed, g2_b, 1,
                  src, dst, ea, a_bf, b_bf, hf, bufB, a_s, a_d, mkey, zz, llog, ebuf,
                  easum, weff, aeloop, stream);
    run_gat_layer(bufB, HEADS * HID, 1, HID, g3_w, g3_asrc, g3_adst, g3_we, g3_aed, g3_b, 0,
                  src, dst, ea, a_bf, b_bf, hf, bufA, a_s, a_d, mkey, zz, llog, ebuf,
                  easum, weff, aeloop, stream);

    // global mean pool
    zero_kernel<<<1, 256, 0, stream>>>(gsum, (size_t)N_GRAPHS * HID);
    zero_kernel<<<1, 256, 0, stream>>>(cnt, N_GRAPHS);
    pool_sum_kernel<<<(N_NODES * HID + 255) / 256, 256, 0, stream>>>(bufA, batch, gsum);
    pool_cnt_kernel<<<(N_NODES + 255) / 256, 256, 0, stream>>>(batch, cnt);
    pool_div_kernel<<<(N_GRAPHS * HID + 255) / 256, 256, 0, stream>>>(gsum, cnt, g);

    // projection + vulnerability heads
    proj1_kernel<<<(N_GRAPHS * 2 * HID + 255) / 256, 256, 0, stream>>>(g, proj_w1, proj_b1, p1);
    proj2_kernel<<<(N_GRAPHS * OUT_DIM + 255) / 256, 256, 0, stream>>>(p1, proj_w2, proj_b2, statf);
    vuln_kernel<<<1, 64, 0, stream>>>(statf, vuln_w, vuln_b, scores);
}
